// EnergyCAModel_19507741458730
// MI455X (gfx1250) — matrix operand order repaired and checked by reading, not compiled, not run on hardware
//
#include <hip/hip_runtime.h>
#include <cstdint>

#define B_    16
#define H_    128
#define W_    128
#define C_    16
#define HID_  256
#define K3C   48
#define KPAD  64
#define STEPS 4
#define NPIX  (B_*H_*W_)

typedef __attribute__((ext_vector_type(16))) _Float16 v16h;
typedef __attribute__((ext_vector_type(8)))  float    v8f;

union Frag { v16h h; uint32_t u[8]; };

__device__ __forceinline__ float wang_u01(uint32_t x){
  x = (x ^ 61u) ^ (x >> 16);
  x *= 9u;
  x ^= x >> 4;
  x *= 0x27d4eb2du;
  x ^= x >> 15;
  return (float)(x & 0xFFFFFFu) * (1.0f/16777216.0f);
}

// ---------------------------------------------------------------------------
// Per-pixel Sobel features (f16) + pre-alive mask (3x3 maxpool of ch3 > 0.1)
// ---------------------------------------------------------------------------
__global__ void feat_kernel(const float* __restrict__ x,
                            _Float16* __restrict__ dxb,
                            uint8_t* __restrict__ preb){
  int p = blockIdx.x * blockDim.x + threadIdx.x;
  if (p >= NPIX) return;
  int b  = p / (H_*W_);
  int hw = p % (H_*W_);
  int h  = hw / W_, w = hw % W_;
  const float* xb = x + (size_t)b * H_*W_*C_;
  _Float16* drow = dxb + (size_t)p * K3C;
  float ch3max = -1e30f;

  for (int c = 0; c < C_; ++c){
    float n[3][3];
    #pragma unroll
    for (int i = 0; i < 3; ++i){
      #pragma unroll
      for (int j = 0; j < 3; ++j){
        int hh = h + i - 1, ww = w + j - 1;
        float v = 0.0f;
        if (hh >= 0 && hh < H_ && ww >= 0 && ww < W_)
          v = xb[((size_t)hh * W_ + ww) * C_ + c];
        n[i][j] = v;
      }
    }
    if (c == 3){
      float m = n[0][0];
      #pragma unroll
      for (int i = 0; i < 3; ++i)
        #pragma unroll
        for (int j = 0; j < 3; ++j)
          m = m > n[i][j] ? m : n[i][j];
      ch3max = m;
    }
    float gx = ((n[0][2]-n[0][0]) + 2.0f*(n[1][2]-n[1][0]) + (n[2][2]-n[2][0])) * 0.125f;
    float gy = ((n[2][0]-n[0][0]) + 2.0f*(n[2][1]-n[0][1]) + (n[2][2]-n[0][2])) * 0.125f;
    drow[c]        = (_Float16)n[1][1];
    drow[16 + c]   = (_Float16)gx;
    drow[32 + c]   = (_Float16)gy;
  }
  preb[p] = (ch3max > 0.1f) ? 1 : 0;
}

// ---------------------------------------------------------------------------
// Weight conversion to f16.  w0h: [HID][KPAD] (K zero-padded 48->64),
// w1h: [C][HID].  Both layouts give contiguous (k,k+1) pairs per B-lane.
// ---------------------------------------------------------------------------
__global__ void cvt_kernel(const float* __restrict__ w0,
                           const float* __restrict__ w1,
                           _Float16* __restrict__ w0h,
                           _Float16* __restrict__ w1h){
  int t = blockIdx.x * blockDim.x + threadIdx.x;
  if (t < HID_ * KPAD){
    int n = t / KPAD, k = t % KPAD;
    w0h[t] = (k < K3C) ? (_Float16)w0[n * K3C + k] : (_Float16)0.0f;
  }
  if (t < C_ * HID_){
    w1h[t] = (_Float16)w1[t];
  }
}

// ---------------------------------------------------------------------------
// Fused MLP: dx[64x48] @ w0^T -> relu -> @ w1^T, fire-rate head, gumbel gate,
// residual add.  256 threads = 8 waves per 64-pixel tile.
// A tile staged into LDS with CDNA5 async DMA (GLOBAL_LOAD_ASYNC_TO_LDS_B128,
// tracked on ASYNCcnt), bypassing VGPRs.
// ---------------------------------------------------------------------------
__global__ void __launch_bounds__(256) mlp_kernel(
    const _Float16* __restrict__ dxb,
    const _Float16* __restrict__ w0h,   // [HID][KPAD]
    const _Float16* __restrict__ w1h,   // [C][HID]
    const float* __restrict__ b0,
    const float* __restrict__ wfr,
    const uint8_t* __restrict__ preb,
    const float* __restrict__ xcur,
    float* __restrict__ xnext,
    float* __restrict__ out_fr,
    int step)
{
  __shared__ alignas(16) _Float16 sA[64 * KPAD];   // dx tile, K zero-padded
  __shared__ alignas(16) _Float16 sH[64 * HID_];   // relu(hidden), staged for GEMM2
  __shared__ float sGate[64];

  const int tid  = threadIdx.x;
  const int lane = tid & 31;
  const int wave = tid >> 5;
  const int base = blockIdx.x * 64;

  __builtin_prefetch(w0h, 0, 3);   // global_prefetch_b8: warm weights into cache

  // ---- stage A tile: async DMA global->LDS, 16B per lane per issue ----
  {
    // 64 rows x 96B of features = 384 x 16B chunks, 256 threads
    for (int c = tid; c < 384; c += 256){
      int row = c / 6, seg = c % 6;
      uint32_t lds = (uint32_t)(uintptr_t)(sA + row * KPAD) + (uint32_t)seg * 16u;
      uint32_t go  = (uint32_t)(base + row) * (uint32_t)(K3C * 2) + (uint32_t)seg * 16u;
      asm volatile("global_load_async_to_lds_b128 %0, %1, %2"
                   :: "v"(lds), "v"(go), "s"(dxb) : "memory");
    }
    // zero the K padding (halves 48..63 of each row) while the DMA runs
    if (tid < 64){
      uint4 z = {0u, 0u, 0u, 0u};
      uint4* dst = (uint4*)(sA + tid * KPAD + K3C);
      dst[0] = z; dst[1] = z;
    }
    asm volatile("s_wait_asynccnt 0x0" ::: "memory");
  }
  __syncthreads();

  // ---- GEMM1: each wave: one 16-row M-strip x 128 N-cols (8 n-tiles, 2 K-steps) ----
  {
    const int msub  = wave >> 1;        // 0..3
    const int nhalf = wave & 1;         // 0..1
    const int mrow  = lane & 15;
    const int hi    = lane >> 4;        // 0 or 1
    for (int nt = 0; nt < 8; ++nt){
      const int n0   = nhalf * 128 + nt * 16;
      const int ncol = n0 + (lane & 15);
      const float bias = b0[ncol];
      v8f acc = {};
      #pragma unroll
      for (int ks = 0; ks < 2; ++ks){
        Frag a, bf;
        #pragma unroll
        for (int v = 0; v < 8; ++v){
          // A 16x32 f16 lane map: lanes<16 hold K {0..7,16..23}, lanes>=16 +8
          int ka = ((v < 4) ? (v * 2) : (16 + (v - 4) * 2)) + hi * 8 + ks * 32;
          a.u[v] = *(const uint32_t*)(sA + (msub * 16 + mrow) * KPAD + ka);
          // B 32x16 f16 lane map: same as A, lanes<16 hold K {0..7,16..23}, lanes>=16 +8
          int kb = ((v < 4) ? (v * 2) : (16 + (v - 4) * 2)) + hi * 8 + ks * 32;
          bf.u[v] = *(const uint32_t*)(w0h + (size_t)ncol * KPAD + kb);
        }
        acc = __builtin_amdgcn_wmma_f32_16x16x32_f16(false, a.h, false, bf.h,
                                                     (short)0, acc, false, false);
      }
      // bias + relu, re-stripe C-layout -> row-major sH
      #pragma unroll
      for (int v = 0; v < 8; ++v){
        float t = acc[v] + bias;
        t = t > 0.0f ? t : 0.0f;
        int M = v + hi * 8;
        sH[(msub * 16 + M) * HID_ + ncol] = (_Float16)t;
      }
    }
  }
  __syncthreads();

  // ---- waves 4..5: fire-rate head + gumbel hard gate for the 64 pixels ----
  if (wave >= 4){
    int pix = (wave - 4) * 32 + lane;
    if (pix < 64){
      int p = base + pix;
      float t = 0.0f;
      for (int k = 0; k < K3C; ++k)
        t += (float)sA[pix * KPAD + k] * wfr[k];
      float fr = 1.0f / (1.0f + __expf(-t));
      fr *= (float)preb[p];
      uint32_t seed = (uint32_t)p * 2654435761u + (uint32_t)(step + 1) * 0x9E3779B9u;
      float u0 = wang_u01(seed);
      float u1 = wang_u01(seed ^ 0x85ebca6bu);
      float g0 = -__logf(-__logf(u0 + 1e-20f) + 1e-20f);
      float g1 = -__logf(-__logf(u1 + 1e-20f) + 1e-20f);
      float s0 = __logf(fr + 1e-10f) + g0;
      float s1 = __logf(1.0f - fr + 1e-10f) + g1;
      sGate[pix] = (s0 > s1) ? 1.0f : 0.0f;
      out_fr[p]  = fr;
    }
  }

  // ---- waves 0..3: GEMM2: sH[16x256] @ w1^T[256x16] ----
  v8f acc2 = {};
  const int msub2 = wave;
  if (wave < 4){
    const int mrow = lane & 15;
    const int hi   = lane >> 4;
    const int ncol = lane & 15;
    #pragma unroll
    for (int ks = 0; ks < 8; ++ks){
      Frag a, bf;
      #pragma unroll
      for (int v = 0; v < 8; ++v){
        int ka = ((v < 4) ? (v * 2) : (16 + (v - 4) * 2)) + hi * 8 + ks * 32;
        a.u[v] = *(const uint32_t*)(sH + (msub2 * 16 + mrow) * HID_ + ka);
        int kb = ((v < 4) ? (v * 2) : (16 + (v - 4) * 2)) + hi * 8 + ks * 32;
        bf.u[v] = *(const uint32_t*)(w1h + (size_t)ncol * HID_ + kb);
      }
      acc2 = __builtin_amdgcn_wmma_f32_16x16x32_f16(false, a.h, false, bf.h,
                                                    (short)0, acc2, false, false);
    }
  }
  __syncthreads();

  // ---- gated residual add ----
  if (wave < 4){
    const int hi   = lane >> 4;
    const int ncol = lane & 15;
    #pragma unroll
    for (int v = 0; v < 8; ++v){
      int M   = v + hi * 8;
      int pix = msub2 * 16 + M;
      size_t idx = (size_t)(base + pix) * C_ + ncol;
      xnext[idx] = xcur[idx] + acc2[v] * sGate[pix];
    }
  }
}

// ---------------------------------------------------------------------------
// Post-alive mask: x *= (pre & post); write step output + next-step state
// ---------------------------------------------------------------------------
__global__ void alive_kernel(const float* __restrict__ xnext,
                             const uint8_t* __restrict__ preb,
                             float* __restrict__ xcur,
                             float* __restrict__ outx){
  int p = blockIdx.x * blockDim.x + threadIdx.x;
  if (p >= NPIX) return;
  int b  = p / (H_*W_);
  int hw = p % (H_*W_);
  int h  = hw / W_, w = hw % W_;
  const float* xb = xnext + (size_t)b * H_*W_*C_;
  float m = -1e30f;
  #pragma unroll
  for (int i = -1; i <= 1; ++i)
    #pragma unroll
    for (int j = -1; j <= 1; ++j){
      int hh = h + i, ww = w + j;
      if (hh >= 0 && hh < H_ && ww >= 0 && ww < W_){
        float v = xb[((size_t)hh * W_ + ww) * C_ + 3];
        m = m > v ? m : v;
      }
    }
  float keep = (m > 0.1f && preb[p]) ? 1.0f : 0.0f;
  for (int c = 0; c < C_; ++c){
    float v = xnext[(size_t)p * C_ + c] * keep;
    xcur[(size_t)p * C_ + c] = v;
    outx[(size_t)p * C_ + c] = v;
  }
}

extern "C" void kernel_launch(void* const* d_in, const int* in_sizes, int n_in,
                              void* d_out, int out_size, void* d_ws, size_t ws_size,
                              hipStream_t stream){
  const float* x   = (const float*)d_in[0];
  const float* w0  = (const float*)d_in[1];
  const float* b0  = (const float*)d_in[2];
  const float* w1  = (const float*)d_in[3];
  const float* wfr = (const float*)d_in[4];
  // steps fixed at 4 per setup_inputs (d_in[5] is a device scalar; unused)

  char* ws = (char*)d_ws;
  size_t off = 0;
  auto alloc = [&](size_t bytes) -> char* {
    char* p = ws + off;
    off += (bytes + 255) & ~(size_t)255;
    return p;
  };
  float*    xcur  = (float*)   alloc((size_t)NPIX * C_ * 4);
  float*    xnext = (float*)   alloc((size_t)NPIX * C_ * 4);
  _Float16* dxb   = (_Float16*)alloc((size_t)NPIX * K3C * 2);
  uint8_t*  preb  = (uint8_t*) alloc((size_t)NPIX);
  _Float16* w0h   = (_Float16*)alloc((size_t)HID_ * KPAD * 2);
  _Float16* w1h   = (_Float16*)alloc((size_t)C_ * HID_ * 2);

  float* out_x  = (float*)d_out;
  float* out_fr = out_x + (size_t)STEPS * NPIX * C_;

  hipMemcpyAsync(xcur, x, (size_t)NPIX * C_ * 4, hipMemcpyDeviceToDevice, stream);
  cvt_kernel<<<(HID_ * KPAD + 255) / 256, 256, 0, stream>>>(w0, w1, w0h, w1h);

  for (int step = 0; step < STEPS; ++step){
    feat_kernel<<<NPIX / 256, 256, 0, stream>>>(xcur, dxb, preb);
    mlp_kernel<<<NPIX / 64, 256, 0, stream>>>(dxb, w0h, w1h, b0, wfr, preb,
                                              xcur, xnext,
                                              out_fr + (size_t)step * NPIX, step);
    alive_kernel<<<NPIX / 256, 256, 0, stream>>>(xnext, preb, xcur,
                                                 out_x + (size_t)step * NPIX * C_);
  }
}
